// RGCN_LP_25606595019029
// MI455X (gfx1250) — matrix operand order repaired and checked by reading, not compiled, not run on hardware
//
#include <hip/hip_runtime.h>
#include <hip/hip_bf16.h>

typedef __attribute__((ext_vector_type(16))) __bf16 v16bf;
typedef __attribute__((ext_vector_type(8)))  float  v8f;

#define ASTR 40   // 32 K-elements + 8 pad (rows 80B -> 16B aligned vector ops)

__device__ __forceinline__ void atomAddF(float* p, float v) {
    unsafeAtomicAdd(p, v);   // native global_atomic_add_f32 on gfx1250
}

// ---------------- conversion kernels ----------------
__global__ void cvt_f32_bf16(const float* __restrict__ in, __bf16* __restrict__ out, size_t n) {
    size_t i = (size_t)blockIdx.x * blockDim.x + threadIdx.x;
    size_t stride = (size_t)gridDim.x * blockDim.x;
    for (; i < n; i += stride) out[i] = (__bf16)in[i];
}

// W [K][N] f32 -> Wt [N][K] bf16 (pre-transpose so GEMM B-tiles load like A-tiles)
__global__ void cvt_tr_f32_bf16(const float* __restrict__ in, __bf16* __restrict__ out, int K, int N) {
    int idx = blockIdx.x * blockDim.x + threadIdx.x;
    if (idx < K * N) {
        int k = idx / N, n = idx % N;
        out[(size_t)n * K + k] = (__bf16)in[idx];
    }
}

__global__ void relu_cvt_f32_bf16(const float* __restrict__ in, __bf16* __restrict__ out, size_t n) {
    size_t i = (size_t)blockIdx.x * blockDim.x + threadIdx.x;
    size_t stride = (size_t)gridDim.x * blockDim.x;
    for (; i < n; i += stride) out[i] = (__bf16)fmaxf(in[i], 0.0f);
}

__global__ void zero_f32(float* __restrict__ p, size_t n) {
    size_t i = (size_t)blockIdx.x * blockDim.x + threadIdx.x;
    size_t stride = (size_t)gridDim.x * blockDim.x;
    for (; i < n; i += stride) p[i] = 0.0f;
}

// ---------------- graph degree / inverse ----------------
__global__ void deg_count(const int* __restrict__ dst, const int* __restrict__ et,
                          float* __restrict__ deg, int E, int NN) {
    int e = blockIdx.x * blockDim.x + threadIdx.x;
    if (e < E) atomAddF(&deg[(size_t)et[e] * NN + dst[e]], 1.0f);
}

__global__ void inv_deg(float* __restrict__ deg, size_t n) {
    size_t i = (size_t)blockIdx.x * blockDim.x + threadIdx.x;
    if (i < n) deg[i] = 1.0f / fmaxf(deg[i], 1.0f);
}

// ---------------- bf16 WMMA GEMM ----------------
// C[M,N] = A[M,K](bf16, row-major) * Bt[N,K](bf16, row-major == B^T) + bias
// block: 256 threads = 8 waves; block tile 128(M) x 64(N); wave grid 4x2,
// each wave owns a 32x32 tile = 4 f32 accumulators (2 A-frags x 2 B-frags,
// every fragment reused twice -> 4 WMMAs per 8 ds_load_b128).
template<bool OUT_BF16>
__global__ __launch_bounds__(256) void gemm_wmma_bf16(
    const __bf16* __restrict__ A, const __bf16* __restrict__ Bt,
    const float* __restrict__ bias, void* __restrict__ Cp,
    int M, int N, int K)
{
    __shared__ __bf16 sA[128 * ASTR];
    __shared__ __bf16 sB[64 * ASTR];

    const int tid  = threadIdx.x;
    const int m0   = blockIdx.x * 128;
    const int n0   = blockIdx.y * 64;
    const int w    = tid >> 5, lane = tid & 31;
    const int wm   = w & 3,  wn   = w >> 2;     // 4 x 2 wave grid
    const int lm   = lane & 15;
    const int hi   = lane >> 4;                 // lane half

    // staging indices: thread moves 16B vectors (8 bf16)
    const int r  = tid >> 2;                    // 0..63
    const int g8 = (tid & 3) * 8;               // k-group

    v8f acc00 = {0,0,0,0,0,0,0,0};
    v8f acc01 = {0,0,0,0,0,0,0,0};
    v8f acc10 = {0,0,0,0,0,0,0,0};
    v8f acc11 = {0,0,0,0,0,0,0,0};

    union Frag { v16bf v; uint4 q[2]; };

    // clamp out-of-range rows to M-1: unconditional loads, duplicated boundary
    // data in LDS is dead math (those output rows are never stored)
    const int gm0 = min(m0 + r,      M - 1);
    const int gm1 = min(m0 + r + 64, M - 1);

    for (int kk = 0; kk < K; kk += 32) {
        __syncthreads();
        {   // A tile: rows m0..m0+127, cols kk..kk+31
            uint4 va0 = *(const uint4*)(A + (size_t)gm0 * K + kk + g8);
            uint4 va1 = *(const uint4*)(A + (size_t)gm1 * K + kk + g8);
            *(uint4*)(&sA[r * ASTR + g8])        = va0;
            *(uint4*)(&sA[(r + 64) * ASTR + g8]) = va1;
            // B tile (transposed weights): rows n0..n0+63 (N,K are tile-multiples)
            uint4 vb = *(const uint4*)(Bt + (size_t)(n0 + r) * K + kk + g8);
            *(uint4*)(&sB[r * ASTR + g8]) = vb;
        }
        __syncthreads();

        // A fragment: lane holds row lm, K = {hi*8..+7} and {hi*8+16..+23}
        Frag a0, a1, b0, b1;
        const __bf16* pa0 = &sA[(wm * 32 + lm)      * ASTR + hi * 8];
        const __bf16* pa1 = &sA[(wm * 32 + 16 + lm) * ASTR + hi * 8];
        a0.q[0] = *(const uint4*)(pa0);  a0.q[1] = *(const uint4*)(pa0 + 16);
        a1.q[0] = *(const uint4*)(pa1);  a1.q[1] = *(const uint4*)(pa1 + 16);
        // B fragment: lane holds col lm, K = {hi*8..+7} and {hi*8+16..+23} in NxK layout
        const __bf16* pb0 = &sB[(wn * 32 + lm)      * ASTR + hi * 8];
        const __bf16* pb1 = &sB[(wn * 32 + 16 + lm) * ASTR + hi * 8];
        b0.q[0] = *(const uint4*)(pb0);  b0.q[1] = *(const uint4*)(pb0 + 16);
        b1.q[0] = *(const uint4*)(pb1);  b1.q[1] = *(const uint4*)(pb1 + 16);

        acc00 = __builtin_amdgcn_wmma_f32_16x16x32_bf16(false, a0.v, false, b0.v, (short)0, acc00, false, false);
        acc01 = __builtin_amdgcn_wmma_f32_16x16x32_bf16(false, a0.v, false, b1.v, (short)0, acc01, false, false);
        acc10 = __builtin_amdgcn_wmma_f32_16x16x32_bf16(false, a1.v, false, b0.v, (short)0, acc10, false, false);
        acc11 = __builtin_amdgcn_wmma_f32_16x16x32_bf16(false, a1.v, false, b1.v, (short)0, acc11, false, false);
    }

    const int gn0 = n0 + wn * 32 + lm;
    const int gn1 = gn0 + 16;
    const float bv0 = bias ? bias[gn0] : 0.0f;
    const float bv1 = bias ? bias[gn1] : 0.0f;
#pragma unroll
    for (int j = 0; j < 8; ++j) {
        int r0 = m0 + wm * 32 + j + hi * 8;   // rows of acc0*
        int r1 = r0 + 16;                      // rows of acc1*
        if (OUT_BF16) {
            __bf16* C = (__bf16*)Cp;
            if (r0 < M) {
                C[(size_t)r0 * N + gn0] = (__bf16)(acc00[j] + bv0);
                C[(size_t)r0 * N + gn1] = (__bf16)(acc01[j] + bv1);
            }
            if (r1 < M) {
                C[(size_t)r1 * N + gn0] = (__bf16)(acc10[j] + bv0);
                C[(size_t)r1 * N + gn1] = (__bf16)(acc11[j] + bv1);
            }
        } else {
            float* C = (float*)Cp;
            if (r0 < M) {
                C[(size_t)r0 * N + gn0] = acc00[j] + bv0;
                C[(size_t)r0 * N + gn1] = acc01[j] + bv1;
            }
            if (r1 < M) {
                C[(size_t)r1 * N + gn0] = acc10[j] + bv0;
                C[(size_t)r1 * N + gn1] = acc11[j] + bv1;
            }
        }
    }
}

// ---------------- edge scatter (mean fused via inv-degree) ----------------
// one wave per edge: out[dst] += h_r[src] * inv[r][dst]
__global__ __launch_bounds__(256) void scatter_edges(
    const int* __restrict__ src, const int* __restrict__ dst, const int* __restrict__ et,
    const float* __restrict__ h0, const float* __restrict__ h1,
    const float* __restrict__ inv, float* __restrict__ out, int E, int C, int NN)
{
    int w = (blockIdx.x * blockDim.x + threadIdx.x) >> 5;
    int lane = threadIdx.x & 31;
    if (w >= E) return;
    int s = src[w], d = dst[w], r = et[w];
    const float* h = (r == 0 ? h0 : h1) + (size_t)s * C;
    float wt = inv[(size_t)r * NN + d];
    float* o = out + (size_t)d * C;
    for (int c = lane; c < C; c += 32)
        atomAddF(&o[c], h[c] * wt);
}

// ---------------- decode: sigmoid(concat(z[d0],z[d1]) @ fc_w + fc_b) ----------------
__global__ __launch_bounds__(256) void decode_pairs(
    const float* __restrict__ z, const int* __restrict__ d0, const int* __restrict__ d1,
    const float* __restrict__ fw, const float* __restrict__ fb,
    float* __restrict__ out, int P)
{
    int w = (blockIdx.x * blockDim.x + threadIdx.x) >> 5;
    int lane = threadIdx.x & 31;
    if (w >= P) return;
    const float* za = z + (size_t)d0[w] * 128;
    const float* zb = z + (size_t)d1[w] * 128;
    float s = 0.0f;
#pragma unroll
    for (int c = lane; c < 128; c += 32)
        s += za[c] * fw[c] + zb[c] * fw[128 + c];
#pragma unroll
    for (int o = 16; o > 0; o >>= 1) s += __shfl_down(s, o, 32);
    if (lane == 0) out[w] = 1.0f / (1.0f + __expf(-(s + fb[0])));
}

// ---------------- host launcher ----------------
extern "C" void kernel_launch(void* const* d_in, const int* in_sizes, int n_in,
                              void* d_out, int out_size, void* d_ws, size_t ws_size,
                              hipStream_t stream) {
    const float* x0    = (const float*)d_in[0];
    const float* x1    = (const float*)d_in[1];
    const int*   ei    = (const int*)d_in[2];
    const int*   et    = (const int*)d_in[3];
    const int*   dec   = (const int*)d_in[4];
    const float* l0w   = (const float*)d_in[5];
    const float* l0b   = (const float*)d_in[6];
    const float* l1w   = (const float*)d_in[7];
    const float* l1b   = (const float*)d_in[8];
    const float* c1w   = (const float*)d_in[9];
    const float* c1r   = (const float*)d_in[10];
    const float* c1b   = (const float*)d_in[11];
    const float* c2w   = (const float*)d_in[12];
    const float* c2r   = (const float*)d_in[13];
    const float* c2b   = (const float*)d_in[14];
    const float* fcw   = (const float*)d_in[15];
    const float* fcb   = (const float*)d_in[16];

    const int M0 = in_sizes[0] / 128;      // 50000
    const int M1 = in_sizes[1] / 64;       // 50000
    const int NN = M0 + M1;                // 100000
    const int E  = in_sizes[2] / 2;        // 400000
    const int P  = in_sizes[4] / 2;        // 100000
    const int Cin = 128, Chid = 256, Cout = 128;

    // workspace layout
    char* ws = (char*)d_ws;
    size_t off = 0;
    auto take = [&](size_t bytes) { size_t r = off; off = (off + bytes + 255) & ~(size_t)255; return r; };
    __bf16* wb_l0 = (__bf16*)(ws + take((size_t)128 * Cin * 2));
    __bf16* wb_l1 = (__bf16*)(ws + take((size_t)64  * Cin * 2));
    __bf16* wb_r1 = (__bf16*)(ws + take((size_t)Cin * Chid * 2));
    __bf16* wb_c1 = (__bf16*)(ws + take((size_t)2 * Cin * Chid * 2));
    __bf16* wb_r2 = (__bf16*)(ws + take((size_t)Chid * Cout * 2));
    __bf16* wb_c2 = (__bf16*)(ws + take((size_t)2 * Chid * Cout * 2));
    __bf16* xin0  = (__bf16*)(ws + take((size_t)M0 * 128 * 2));
    __bf16* xin1  = (__bf16*)(ws + take((size_t)M1 * 64  * 2));
    __bf16* xb    = (__bf16*)(ws + take((size_t)NN * Cin * 2));
    __bf16* x2b   = (__bf16*)(ws + take((size_t)NN * Chid * 2));
    float*  hA    = (float*) (ws + take((size_t)NN * Chid * 4));
    float*  hB    = (float*) (ws + take((size_t)NN * Chid * 4));
    float*  out1  = (float*) (ws + take((size_t)NN * Chid * 4));
    float*  out2  = (float*) (ws + take((size_t)NN * Cout * 4));
    float*  deg   = (float*) (ws + take((size_t)2 * NN * 4));

    // ---- weight convert+transpose (W[K][N] -> Wt[N][K] bf16) ----
    auto trBlocks = [](int n) { return (n + 255) / 256; };
    cvt_tr_f32_bf16<<<trBlocks(128 * Cin), 256, 0, stream>>>(l0w, wb_l0, 128, Cin);
    cvt_tr_f32_bf16<<<trBlocks(64 * Cin), 256, 0, stream>>>(l1w, wb_l1, 64, Cin);
    cvt_tr_f32_bf16<<<trBlocks(Cin * Chid), 256, 0, stream>>>(c1r, wb_r1, Cin, Chid);
    cvt_tr_f32_bf16<<<trBlocks(Cin * Chid), 256, 0, stream>>>(c1w, wb_c1, Cin, Chid);
    cvt_tr_f32_bf16<<<trBlocks(Cin * Chid), 256, 0, stream>>>(c1w + (size_t)Cin * Chid, wb_c1 + (size_t)Chid * Cin, Cin, Chid);
    cvt_tr_f32_bf16<<<trBlocks(Chid * Cout), 256, 0, stream>>>(c2r, wb_r2, Chid, Cout);
    cvt_tr_f32_bf16<<<trBlocks(Chid * Cout), 256, 0, stream>>>(c2w, wb_c2, Chid, Cout);
    cvt_tr_f32_bf16<<<trBlocks(Chid * Cout), 256, 0, stream>>>(c2w + (size_t)Chid * Cout, wb_c2 + (size_t)Cout * Chid, Chid, Cout);

    // ---- activation convert ----
    cvt_f32_bf16<<<4096, 256, 0, stream>>>(x0, xin0, (size_t)M0 * 128);
    cvt_f32_bf16<<<4096, 256, 0, stream>>>(x1, xin1, (size_t)M1 * 64);

    // ---- degrees -> inverse (shared by both layers) ----
    zero_f32<<<trBlocks(2 * NN), 256, 0, stream>>>(deg, (size_t)2 * NN);
    deg_count<<<(E + 255) / 256, 256, 0, stream>>>(ei + E, et, deg, E, NN);
    inv_deg<<<trBlocks(2 * NN), 256, 0, stream>>>(deg, (size_t)2 * NN);

    // ---- input projections -> xb (bf16) ----
    gemm_wmma_bf16<true><<<dim3((M0 + 127) / 128, Cin / 64), 256, 0, stream>>>(xin0, wb_l0, l0b, xb, M0, Cin, 128);
    gemm_wmma_bf16<true><<<dim3((M1 + 127) / 128, Cin / 64), 256, 0, stream>>>(xin1, wb_l1, l1b, xb + (size_t)M0 * Cin, M1, Cin, 64);

    // ---- layer 1: root + per-relation transforms ----
    dim3 g1((NN + 127) / 128, Chid / 64);
    gemm_wmma_bf16<false><<<g1, 256, 0, stream>>>(xb, wb_r1, c1b, out1, NN, Chid, Cin);
    gemm_wmma_bf16<false><<<g1, 256, 0, stream>>>(xb, wb_c1, nullptr, hA, NN, Chid, Cin);
    gemm_wmma_bf16<false><<<g1, 256, 0, stream>>>(xb, wb_c1 + (size_t)Chid * Cin, nullptr, hB, NN, Chid, Cin);
    scatter_edges<<<(E + 7) / 8, 256, 0, stream>>>(ei, ei + E, et, hA, hB, deg, out1, E, Chid, NN);
    relu_cvt_f32_bf16<<<4096, 256, 0, stream>>>(out1, x2b, (size_t)NN * Chid);

    // ---- layer 2 ----
    dim3 g2((NN + 127) / 128, Cout / 64);
    gemm_wmma_bf16<false><<<g2, 256, 0, stream>>>(x2b, wb_r2, c2b, out2, NN, Cout, Chid);
    gemm_wmma_bf16<false><<<g2, 256, 0, stream>>>(x2b, wb_c2, nullptr, hA, NN, Cout, Chid);
    gemm_wmma_bf16<false><<<g2, 256, 0, stream>>>(x2b, wb_c2 + (size_t)Cout * Chid, nullptr, hB, NN, Cout, Chid);
    scatter_edges<<<(E + 7) / 8, 256, 0, stream>>>(ei, ei + E, et, hA, hB, deg, out2, E, Cout, NN);

    // ---- decode ----
    decode_pairs<<<(P + 7) / 8, 256, 0, stream>>>(out2, dec, dec + P, fcw, fcb, (float*)d_out, P);
}
